// LongFormer_687194767379
// MI455X (gfx1250) — matrix operand order repaired and checked by reading, not compiled, not run on hardware
//
#include <hip/hip_runtime.h>
#include <hip/hip_bf16.h>

// ---------------------------------------------------------------------------
// CDNA5 (gfx1250) Longformer forward.
// bf16 operand pipeline + WMMA f32 accum + TDM (tensor_load_to_lds) staging.
// ---------------------------------------------------------------------------

typedef __bf16        v16bf __attribute__((ext_vector_type(16)));
typedef float         v8f   __attribute__((ext_vector_type(8)));
typedef unsigned int  u32x4 __attribute__((ext_vector_type(4)));
typedef float         f32x4 __attribute__((ext_vector_type(4)));
typedef int           i32x4 __attribute__((ext_vector_type(4)));
typedef int           i32x8 __attribute__((ext_vector_type(8)));

#define S_LEN   4096
#define D_MODEL 768
#define N_HEAD  12
#define D_HEAD  64
#define N_LAYER 12
#define FF_DIM  3072
#define WIN     256

#ifndef __has_builtin
#define __has_builtin(x) 0
#endif
#if __has_builtin(__builtin_amdgcn_tensor_load_to_lds) && \
    __has_builtin(__builtin_amdgcn_s_wait_tensorcnt)
#define HAVE_TDM 1
#else
#define HAVE_TDM 0
#endif

// ----- scalar helpers -------------------------------------------------------

__device__ __forceinline__ unsigned short f2bfu(float f) {
  unsigned u = __builtin_bit_cast(unsigned, f);
  u = u + 0x7FFFu + ((u >> 16) & 1u);   // round-to-nearest-even
  return (unsigned short)(u >> 16);
}
__device__ __forceinline__ v8f vzero8() {
  v8f z = {0.f, 0.f, 0.f, 0.f, 0.f, 0.f, 0.f, 0.f};
  return z;
}
__device__ __forceinline__ v8f wmma_bf16(v16bf a, v16bf b, v8f c) {
  return __builtin_amdgcn_wmma_f32_16x16x32_bf16(
      false, a, false, b, (short)0, c, false, false);
}

// A fragment (16x32 bf16): lane(0-15)=row M, khalf=lane>>4.
// e[j]=K(16*khalf+j). rowp 16B aligned.
__device__ __forceinline__ v16bf load_a_frag(const unsigned short* rowp, int khalf) {
  union { u32x4 q[2]; v16bf v; } u;
  u.q[0] = *reinterpret_cast<const u32x4*>(rowp + khalf * 16);
  u.q[1] = *reinterpret_cast<const u32x4*>(rowp + khalf * 16 + 8);
  return u.v;
}
// B fragment (32x16 bf16): lane col N=lane&15, khalf=lane>>4. e[j]=K(16*khalf+j).
__device__ __forceinline__ v16bf load_b_frag(const unsigned short* colp, int khalf) {
  union { u32x4 q[2]; v16bf v; } u;
  u.q[0] = *reinterpret_cast<const u32x4*>(colp + khalf * 16);
  u.q[1] = *reinterpret_cast<const u32x4*>(colp + khalf * 16 + 8);
  return u.v;
}

// ----- TDM: 2D bf16 tile (tile_rows x 32 elems) global -> LDS ---------------
// LDS pad: 4 DWORDs after every 16 DWORDs -> padded row stride 40 elements.

__device__ __forceinline__ void tdm_load_tile_bf16(
    unsigned lds_byte_off, const unsigned short* gsrc,
    unsigned row_stride_elems, unsigned tile_rows) {
#if HAVE_TDM
  unsigned long long ga = (unsigned long long)(size_t)gsrc;
  u32x4 g0 = {0u, 0u, 0u, 0u};
  g0[0] = 1u;                                   // count=1, user descriptor
  g0[1] = lds_byte_off;                         // LDS dest (bytes)
  g0[2] = (unsigned)ga;                         // global_addr[31:0]
  g0[3] = (unsigned)((ga >> 32) & 0x01FFFFFFu) | (2u << 30);  // [56:32] | type=2
  i32x8 g1 = {0, 0, 0, 0, 0, 0, 0, 0};
  // data_size=1 (2B) | pad_enable | pad_interval=3 (16 DW) | pad_amount=3 (4 DW)
  g1[0] = (int)((1u << 16) | (1u << 20) | (3u << 22) | (3u << 25));
  g1[1] = (int)(32u << 16);                     // tensor_dim0 = 32 (lo16)
  g1[2] = (int)((tile_rows & 0xFFFFu) << 16);   // tensor_dim1 lo16
  g1[3] = (int)(32u << 16);                     // tile_dim0 = 32
  g1[4] = (int)(tile_rows & 0xFFFFu);           // tile_dim1; tile_dim2=0
  g1[5] = (int)row_stride_elems;                // tensor_dim0_stride[31:0]
  i32x4 z4 = {0, 0, 0, 0};
#if __clang_major__ >= 23
  i32x8 z8 = {0, 0, 0, 0, 0, 0, 0, 0};
  __builtin_amdgcn_tensor_load_to_lds(g0, g1, z4, z4, z8, 0);
#else
  __builtin_amdgcn_tensor_load_to_lds(g0, g1, z4, z4, 0);
#endif
#else
  (void)lds_byte_off; (void)gsrc; (void)row_stride_elems; (void)tile_rows;
#endif
}

// ----- GEMM: C = act(A[MxK]bf16 * Bt[NxK]bf16^T + bias) ----------------------
// 128x128 block tile, K-step 32, 8 waves x (64x32) wave tile, double-buffered
// TDM staging. Optional f32 and/or bf16 outputs.

#define TM 128
#define TN 128
#define TK 32
#define LDST 40   // padded LDS row stride (elements)

__global__ __launch_bounds__(256) void gemm_bf16_kernel(
    const unsigned short* __restrict__ A,   // M x K bf16 row-major
    const unsigned short* __restrict__ Bt,  // N x K bf16 row-major (W^T)
    const float* __restrict__ bias,
    float* __restrict__ Cf,                 // nullable
    unsigned short* __restrict__ Cb,        // nullable
    int M, int N, int K, int gelu) {
  __shared__ __align__(16) unsigned short sA[2][TM * LDST];
  __shared__ __align__(16) unsigned short sB[2][TN * LDST];

  const int tid   = threadIdx.x;
  const int lane  = tid & 31;
  const int w     = tid >> 5;
  const int wm    = w >> 2;
  const int wn    = w & 3;
  const int khalf = lane >> 4;
  const int l16   = lane & 15;
  const int bm    = blockIdx.y * TM;
  const int bn    = blockIdx.x * TN;

  v8f acc[4][2];
#pragma unroll
  for (int mi = 0; mi < 4; mi++)
#pragma unroll
    for (int ni = 0; ni < 2; ni++) acc[mi][ni] = vzero8();

#if HAVE_TDM
  if (w == 0) {   // wave-level branch: only wave 0 issues the DMA
    tdm_load_tile_bf16((unsigned)(size_t)&sA[0][0], &A[(size_t)bm * K], K, TM);
    tdm_load_tile_bf16((unsigned)(size_t)&sB[0][0], &Bt[(size_t)bn * K], K, TN);
  }
#endif

  int p = 0;
  for (int kt = 0; kt < K; kt += TK, p ^= 1) {
#if HAVE_TDM
    if (w == 0) __builtin_amdgcn_s_wait_tensorcnt(0);
    __syncthreads();                       // buffer p ready; buffer p^1 free
    if (w == 0 && kt + TK < K) {
      tdm_load_tile_bf16((unsigned)(size_t)&sA[p ^ 1][0],
                         &A[(size_t)bm * K + kt + TK], K, TM);
      tdm_load_tile_bf16((unsigned)(size_t)&sB[p ^ 1][0],
                         &Bt[(size_t)bn * K + kt + TK], K, TN);
    }
#else
    __syncthreads();
#pragma unroll
    for (int i = 0; i < 2; i++) {          // 512 x b128 copies, no conversion
      int idx = tid + i * 256;
      int r = idx >> 2, c8 = (idx & 3) * 8;
      *reinterpret_cast<u32x4*>(&sA[p][r * LDST + c8]) =
          *reinterpret_cast<const u32x4*>(&A[(size_t)(bm + r) * K + kt + c8]);
      *reinterpret_cast<u32x4*>(&sB[p][r * LDST + c8]) =
          *reinterpret_cast<const u32x4*>(&Bt[(size_t)(bn + r) * K + kt + c8]);
    }
    __syncthreads();
#endif

    v16bf af[4], bf[2];
#pragma unroll
    for (int mi = 0; mi < 4; mi++)
      af[mi] = load_a_frag(&sA[p][(wm * 64 + mi * 16 + l16) * LDST], khalf);
#pragma unroll
    for (int ni = 0; ni < 2; ni++)
      bf[ni] = load_b_frag(&sB[p][(wn * 32 + ni * 16 + l16) * LDST], khalf);
#pragma unroll
    for (int mi = 0; mi < 4; mi++)
#pragma unroll
      for (int ni = 0; ni < 2; ni++)
        acc[mi][ni] = wmma_bf16(af[mi], bf[ni], acc[mi][ni]);
  }

#pragma unroll
  for (int mi = 0; mi < 4; mi++) {
#pragma unroll
    for (int ni = 0; ni < 2; ni++) {
      int n = bn + wn * 32 + ni * 16 + l16;
      float bv = bias[n];
#pragma unroll
      for (int r = 0; r < 8; r++) {
        int m = bm + wm * 64 + mi * 16 + r + 8 * khalf;
        float vv = acc[mi][ni][r] + bv;
        if (gelu) vv = 0.5f * vv * (1.0f + erff(vv * 0.70710678118654752f));
        if (Cf) Cf[(size_t)m * N + n] = vv;
        if (Cb) Cb[(size_t)m * N + n] = f2bfu(vv);
      }
    }
  }
}

// ----- Weight transpose + f32->bf16: W[KxN] -> Wt[NxK] bf16 ------------------

__global__ __launch_bounds__(256) void transpose_bf16_kernel(
    const float* __restrict__ src, unsigned short* __restrict__ dst,
    int K, int N) {
  __shared__ unsigned short t[32][33];
  const int tid = threadIdx.x;
  const int k0 = blockIdx.y * 32, n0 = blockIdx.x * 32;
#pragma unroll
  for (int pss = 0; pss < 4; pss++) {
    int r = (tid >> 5) + pss * 8, c = tid & 31;
    t[r][c] = f2bfu(src[(size_t)(k0 + r) * N + n0 + c]);
  }
  __syncthreads();
#pragma unroll
  for (int pss = 0; pss < 4; pss++) {
    int r = (tid >> 5) + pss * 8, c = tid & 31;   // r: n index, c: k index
    dst[(size_t)(n0 + r) * K + k0 + c] = t[c][r];
  }
}

// ----- Sliding-window attention (bf16 q/k/v, bf16 out) -----------------------
// block = (16 query rows, head). Key window [g0-256, g0+288) = 544 keys.

#define KW 544
#define SSTR 560

__global__ __launch_bounds__(256) void attn_kernel(
    const unsigned short* __restrict__ q, const unsigned short* __restrict__ k,
    const unsigned short* __restrict__ v, unsigned short* __restrict__ out) {
  __shared__ __align__(16) unsigned short sQ[16 * 72];
  __shared__ __align__(16) float sS[16 * SSTR];          // exp(score - max)
  __shared__ __align__(16) unsigned short sP[16 * SSTR]; // probs (bf16)
  __shared__ float sO[8 * 256];

  const int tid   = threadIdx.x;
  const int lane  = tid & 31;
  const int w     = tid >> 5;
  const int khalf = lane >> 4;
  const int l16   = lane & 15;
  const int g0    = blockIdx.x * 16;
  const int h     = blockIdx.y;
  const size_t hoff = (size_t)h * D_HEAD;

  // stage Q (16x64 bf16): raw b128 copies
  if (tid < 128) {
    int r = tid >> 3, c8 = (tid & 7) * 8;
    *reinterpret_cast<u32x4*>(&sQ[r * 72 + c8]) =
        *reinterpret_cast<const u32x4*>(&q[(size_t)(g0 + r) * D_MODEL + hoff + c8]);
  }
  __syncthreads();

  // ---- phase 1: scores = Q K^T / 8, band-masked ---------------------------
  for (int kt = w; kt < 34; kt += 8) {
    const int key = g0 - WIN + kt * 16 + l16;
    const bool kok = (key >= 0) && (key < S_LEN);
    v8f acc = vzero8();
#pragma unroll
    for (int ks = 0; ks < 2; ks++) {
      const int dh0 = ks * 32 + khalf * 16;
      union { u32x4 q2[2]; v16bf v; } ub;
      if (kok) {
        const u32x4* kp =
            reinterpret_cast<const u32x4*>(&k[(size_t)key * D_MODEL + hoff + dh0]);
        ub.q2[0] = kp[0]; ub.q2[1] = kp[1];
      } else {
        u32x4 z = {0u, 0u, 0u, 0u};
        ub.q2[0] = z; ub.q2[1] = z;
      }
      v16bf afr = load_a_frag(&sQ[l16 * 72 + ks * 32], khalf);
      acc = wmma_bf16(afr, ub.v, acc);
    }
    const int jc   = kt * 16 + l16;
    const int keyg = g0 - WIN + jc;
#pragma unroll
    for (int r = 0; r < 8; r++) {
      int rl = r + 8 * khalf;
      int d  = keyg - (g0 + rl);
      bool ok = (keyg >= 0) && (keyg < S_LEN) && (d >= -WIN) && (d <= WIN);
      sS[rl * SSTR + jc] = ok ? acc[r] * 0.125f : -1.0e9f;
    }
  }
  __syncthreads();

  // ---- phase 2: softmax (2 rows per wave); write probs as bf16 -------------
#pragma unroll
  for (int rr = 0; rr < 2; rr++) {
    const int row = w * 2 + rr;
    float* sr = &sS[row * SSTR];
    float m = -3.4e38f;
    for (int j = lane; j < KW; j += 32) m = fmaxf(m, sr[j]);
#pragma unroll
    for (int o = 16; o; o >>= 1) m = fmaxf(m, __shfl_xor(m, o, 32));
    float s = 0.0f;
    for (int j = lane; j < KW; j += 32) { float e = __expf(sr[j] - m); sr[j] = e; s += e; }
#pragma unroll
    for (int o = 16; o; o >>= 1) s += __shfl_xor(s, o, 32);
    float inv = 1.0f / s;
    for (int j = lane; j < KW; j += 32) sP[row * SSTR + j] = f2bfu(sr[j] * inv);
  }
  __syncthreads();

  // ---- phase 3: out = P(16x544) V(544x64), K-split across wave pairs ------
  {
    const int ni = w >> 1;
    const int hk = w & 1;
    v8f acc = vzero8();
    for (int ks = hk; ks < 17; ks += 2) {
      v16bf afr = load_a_frag(&sP[l16 * SSTR + ks * 32], khalf);
      v16bf bfrag;
      const int dh = ni * 16 + l16;
      const int keyb = g0 - WIN + ks * 32 + khalf * 16;
#pragma unroll
      for (int j = 0; j < 16; j++) {
        int key = keyb + j;
        unsigned short uv = (key >= 0 && key < S_LEN)
                                ? v[(size_t)key * D_MODEL + hoff + dh]
                                : (unsigned short)0;
        bfrag[j] = __builtin_bit_cast(__bf16, uv);
      }
      acc = wmma_bf16(afr, bfrag, acc);
    }
#pragma unroll
    for (int r = 0; r < 8; r++)
      sO[w * 256 + (r + 8 * khalf) * 16 + l16] = acc[r];
  }
  __syncthreads();
#pragma unroll
  for (int i = 0; i < 4; i++) {
    int idx = tid + i * 256;
    int r = idx >> 6, dh = idx & 63;
    int ni = dh >> 4, c = dh & 15;
    float vv = sO[(ni * 2 + 0) * 256 + r * 16 + c] + sO[(ni * 2 + 1) * 256 + r * 16 + c];
    out[(size_t)(g0 + r) * D_MODEL + hoff + dh] = f2bfu(vv);
  }
}

// ----- Embedding + LayerNorm (f32 + bf16 outputs) ----------------------------

__global__ __launch_bounds__(256) void embed_ln_kernel(
    const int* __restrict__ x, const float* __restrict__ we,
    const float* __restrict__ pe, const float* __restrict__ tt,
    const float* __restrict__ g, const float* __restrict__ b,
    float* __restrict__ h, unsigned short* __restrict__ hb) {
  __shared__ float red[16];
  const int s = blockIdx.x, tid = threadIdx.x;
  const size_t id = (size_t)x[s];
  float xv[3];
#pragma unroll
  for (int i = 0; i < 3; i++) {
    int c = tid + i * 256;
    xv[i] = we[id * D_MODEL + c] + pe[(size_t)(s + 2) * D_MODEL + c] + tt[c];
  }
  float sum = xv[0] + xv[1] + xv[2];
  float sq  = xv[0]*xv[0] + xv[1]*xv[1] + xv[2]*xv[2];
#pragma unroll
  for (int o = 16; o; o >>= 1) { sum += __shfl_xor(sum, o, 32); sq += __shfl_xor(sq, o, 32); }
  int wv = tid >> 5, lane = tid & 31;
  if (lane == 0) { red[wv] = sum; red[8 + wv] = sq; }
  __syncthreads();
  if (tid == 0) {
    float S = 0.f, Q = 0.f;
    for (int i = 0; i < 8; i++) { S += red[i]; Q += red[8 + i]; }
    float mu = S / (float)D_MODEL;
    red[0] = mu; red[1] = Q / (float)D_MODEL - mu * mu;
  }
  __syncthreads();
  float mu = red[0], inv = rsqrtf(red[1] + 1e-5f);
#pragma unroll
  for (int i = 0; i < 3; i++) {
    int c = tid + i * 256;
    float vv = (xv[i] - mu) * inv * g[c] + b[c];
    h[(size_t)s * D_MODEL + c] = vv;
    hb[(size_t)s * D_MODEL + c] = f2bfu(vv);
  }
}

// ----- Residual add + LayerNorm (in-place on h; bf16 copy) -------------------

__global__ __launch_bounds__(256) void add_ln_kernel(
    float* __restrict__ h, const float* __restrict__ t,
    const float* __restrict__ g, const float* __restrict__ b,
    unsigned short* __restrict__ hb) {
  __shared__ float red[16];
  const int s = blockIdx.x, tid = threadIdx.x;
  float xv[3];
#pragma unroll
  for (int i = 0; i < 3; i++) {
    int c = tid + i * 256;
    xv[i] = h[(size_t)s * D_MODEL + c] + t[(size_t)s * D_MODEL + c];
  }
  float sum = xv[0] + xv[1] + xv[2];
  float sq  = xv[0]*xv[0] + xv[1]*xv[1] + xv[2]*xv[2];
#pragma unroll
  for (int o = 16; o; o >>= 1) { sum += __shfl_xor(sum, o, 32); sq += __shfl_xor(sq, o, 32); }
  int wv = tid >> 5, lane = tid & 31;
  if (lane == 0) { red[wv] = sum; red[8 + wv] = sq; }
  __syncthreads();
  if (tid == 0) {
    float S = 0.f, Q = 0.f;
    for (int i = 0; i < 8; i++) { S += red[i]; Q += red[8 + i]; }
    float mu = S / (float)D_MODEL;
    red[0] = mu; red[1] = Q / (float)D_MODEL - mu * mu;
  }
  __syncthreads();
  float mu = red[0], inv = rsqrtf(red[1] + 1e-5f);
#pragma unroll
  for (int i = 0; i < 3; i++) {
    int c = tid + i * 256;
    float vv = (xv[i] - mu) * inv * g[c] + b[c];
    h[(size_t)s * D_MODEL + c] = vv;
    hb[(size_t)s * D_MODEL + c] = f2bfu(vv);
  }
}

// ----- Final projection: out = h[0,:] @ out_w + out_b ------------------------

__global__ __launch_bounds__(256) void outproj_kernel(
    const float* __restrict__ h, const float* __restrict__ wt,
    const float* __restrict__ b, float* __restrict__ o) {
  int c = threadIdx.x;
  float s = b[c];
  for (int kk = 0; kk < D_MODEL; kk++) s += h[kk] * wt[(size_t)kk * 256 + c];
  o[c] = s;
}

// ----- Orchestration ---------------------------------------------------------

extern "C" void kernel_launch(void* const* d_in, const int* in_sizes, int n_in,
                              void* d_out, int out_size, void* d_ws, size_t ws_size,
                              hipStream_t stream) {
  (void)in_sizes; (void)n_in; (void)out_size; (void)ws_size;

  const int*   x   = (const int*)  d_in[0];
  const float* we  = (const float*)d_in[1];
  const float* pe  = (const float*)d_in[2];
  const float* tt  = (const float*)d_in[3];
  const float* elg = (const float*)d_in[4];
  const float* elb = (const float*)d_in[5];
  const float* Wq  = (const float*)d_in[6];
  const float* bq  = (const float*)d_in[7];
  const float* Wk  = (const float*)d_in[8];
  const float* bk  = (const float*)d_in[9];
  const float* Wv  = (const float*)d_in[10];
  const float* bv  = (const float*)d_in[11];
  const float* Wo  = (const float*)d_in[12];
  const float* bo  = (const float*)d_in[13];
  const float* g1  = (const float*)d_in[14];
  const float* b1  = (const float*)d_in[15];
  const float* Wi  = (const float*)d_in[16];
  const float* bi  = (const float*)d_in[17];
  const float* Wd  = (const float*)d_in[18];
  const float* bd  = (const float*)d_in[19];
  const float* g2  = (const float*)d_in[20];
  const float* b2  = (const float*)d_in[21];
  const float* ow  = (const float*)d_in[22];
  const float* ob  = (const float*)d_in[23];

  const size_t SD = (size_t)S_LEN * D_MODEL;
  const size_t SF = (size_t)S_LEN * FF_DIM;
  const size_t DD = (size_t)D_MODEL * D_MODEL;
  const size_t DF = (size_t)D_MODEL * FF_DIM;

  char* base = (char*)d_ws;
  float* h  = (float*)base;                 base += SD * 4;
  float* tb = (float*)base;                 base += SD * 4;
  unsigned short* hb  = (unsigned short*)base; base += SD * 2;
  unsigned short* qb  = (unsigned short*)base; base += SD * 2;
  unsigned short* kb  = (unsigned short*)base; base += SD * 2;
  unsigned short* vb  = (unsigned short*)base; base += SD * 2;
  unsigned short* ab  = (unsigned short*)base; base += SD * 2;
  unsigned short* fb  = (unsigned short*)base; base += SF * 2;
  unsigned short* wqt = (unsigned short*)base; base += DD * 2;
  unsigned short* wkt = (unsigned short*)base; base += DD * 2;
  unsigned short* wvt = (unsigned short*)base; base += DD * 2;
  unsigned short* wot = (unsigned short*)base; base += DD * 2;
  unsigned short* wit = (unsigned short*)base; base += DF * 2;
  unsigned short* wdt = (unsigned short*)base; base += DF * 2;

  embed_ln_kernel<<<S_LEN, 256, 0, stream>>>(x, we, pe, tt, elg, elb, h, hb);

  const dim3 gD (D_MODEL / TN, S_LEN / TM);   // (6, 32)
  const dim3 gFF(FF_DIM  / TN, S_LEN / TM);   // (24, 32)
  const dim3 gAt(S_LEN / 16, N_HEAD);         // (256, 12)
  const dim3 tDD(D_MODEL / 32, D_MODEL / 32); // (24, 24)
  const dim3 tDF(FF_DIM  / 32, D_MODEL / 32); // (96, 24)
  const dim3 tFD(D_MODEL / 32, FF_DIM  / 32); // (24, 96)

  for (int l = 0; l < N_LAYER; l++) {
    const size_t wofs = (size_t)l * DD;
    const size_t bofs = (size_t)l * D_MODEL;

    transpose_bf16_kernel<<<tDD, 256, 0, stream>>>(Wq + wofs, wqt, D_MODEL, D_MODEL);
    transpose_bf16_kernel<<<tDD, 256, 0, stream>>>(Wk + wofs, wkt, D_MODEL, D_MODEL);
    transpose_bf16_kernel<<<tDD, 256, 0, stream>>>(Wv + wofs, wvt, D_MODEL, D_MODEL);
    transpose_bf16_kernel<<<tDD, 256, 0, stream>>>(Wo + wofs, wot, D_MODEL, D_MODEL);
    transpose_bf16_kernel<<<tDF, 256, 0, stream>>>(Wi + (size_t)l * DF, wit, D_MODEL, FF_DIM);
    transpose_bf16_kernel<<<tFD, 256, 0, stream>>>(Wd + (size_t)l * DF, wdt, FF_DIM, D_MODEL);

    gemm_bf16_kernel<<<gD, 256, 0, stream>>>(hb, wqt, bq + bofs, nullptr, qb,
                                             S_LEN, D_MODEL, D_MODEL, 0);
    gemm_bf16_kernel<<<gD, 256, 0, stream>>>(hb, wkt, bk + bofs, nullptr, kb,
                                             S_LEN, D_MODEL, D_MODEL, 0);
    gemm_bf16_kernel<<<gD, 256, 0, stream>>>(hb, wvt, bv + bofs, nullptr, vb,
                                             S_LEN, D_MODEL, D_MODEL, 0);

    attn_kernel<<<gAt, 256, 0, stream>>>(qb, kb, vb, ab);

    gemm_bf16_kernel<<<gD, 256, 0, stream>>>(ab, wot, bo + bofs, tb, nullptr,
                                             S_LEN, D_MODEL, D_MODEL, 0);
    add_ln_kernel<<<S_LEN, 256, 0, stream>>>(h, tb, g1 + bofs, b1 + bofs, hb);

    gemm_bf16_kernel<<<gFF, 256, 0, stream>>>(hb, wit, bi + (size_t)l * FF_DIM,
                                              nullptr, fb, S_LEN, FF_DIM, D_MODEL, 1);
    gemm_bf16_kernel<<<gD, 256, 0, stream>>>(fb, wdt, bd + bofs, tb, nullptr,
                                             S_LEN, D_MODEL, FF_DIM, 0);
    add_ln_kernel<<<S_LEN, 256, 0, stream>>>(h, tb, g2 + bofs, b2 + bofs, hb);
  }

  outproj_kernel<<<1, 256, 0, stream>>>(h, ow, ob, (float*)d_out);
}
